// PyTorchModel_76965813944525
// MI455X (gfx1250) — compile-verified, matrix operand order checked by reading, not run on hardware
//
#include <hip/hip_runtime.h>
#include <hip/hip_bf16.h>

#define V_DIM 32000
#define E_DIM 512
#define H_DIM 1024
#define S_LEN 2048
#define T_LEN 2048

#define NCHUNK 8
#define VCHUNK (V_DIM / NCHUNK)   // 4000 columns per chunk
#define NTILES (VCHUNK / 16)      // 250 n-tiles per chunk
#define KTILE 512                 // staged K half
#define WROW  (KTILE + 8)         // LDS row pitch (bf16): 1024B + 16B pad (bank spread)
#define RNN_BLOCKS 16

typedef __bf16 bf16;
typedef __attribute__((ext_vector_type(16))) __bf16 v16bf;
typedef __attribute__((ext_vector_type(8)))  float  v8f;
typedef __attribute__((ext_vector_type(4)))  unsigned uint32x4;
typedef __attribute__((ext_vector_type(8)))  int      int32x8;
typedef __attribute__((ext_vector_type(4)))  int      int32x4;

// therock toolchain (6-arg TDM builtin) uniquely ships this header
#if __has_include(<hip/amd_detail/amd_gfx1250_TDM.h>)
#define TDM_6ARG 1
#else
#define TDM_6ARG 0
#endif

// K offset within a 16x32 bf16 A/B fragment (ISA 7.12.2, wave32)
__device__ __forceinline__ int frag_koff(int e, int half) {
    int v = e >> 1;
    int k = (v < 4) ? (2 * v) : (16 + 2 * (v - 4));
    return k + (half << 3) + (e & 1);
}

// Async global->LDS 128-bit copy (CDNA5, tracked with ASYNCcnt)
__device__ __forceinline__ void async_copy_b128(unsigned lds_off, const void* g) {
    asm volatile("global_load_async_to_lds_b128 %0, %1, off"
                 :: "v"(lds_off), "v"((unsigned long long)(uintptr_t)g)
                 : "memory");
}
__device__ __forceinline__ void wait_async_0() {
    asm volatile("s_wait_asynccnt 0x0" ::: "memory");
}

// ---------------------------------------------------------------------------
// TDM: load one 16-row x KTILE-col bf16 tile (row stride H_DIM elems) into LDS
// with 16B padding per row (pad_interval=7 -> every 256 DWORDs, pad_amount=3
// -> 4 DWORDs), reproducing the WROW pitch. D# packed per ISA 8.3/8.4.
// Issued by one wave; tracked with TENSORcnt.
// ---------------------------------------------------------------------------
__device__ __forceinline__ void tdm_load_tile(const bf16* gaddr, unsigned lds_addr) {
    unsigned long long ga = (unsigned long long)(uintptr_t)gaddr;
    uint32x4 g0 = {};
    g0[0] = 1u;                                          // count=1, user mode
    g0[1] = lds_addr;                                    // lds_addr [63:32]
    g0[2] = (unsigned)(ga & 0xFFFFFFFFu);                // global_addr low
    g0[3] = (unsigned)((ga >> 32) & 0x1FFFFFFu) | (2u << 30);  // addr hi | type=2
    int32x8 g1 = {};
    g1[0] = (int)((1u << 16) | (1u << 20) | (7u << 22) | (3u << 25));
    //            data_size=2B  pad_en      pad_int=256dw  pad_amt=4dw
    g1[1] = (int)(((unsigned)H_DIM & 0xFFFFu) << 16);    // tensor_dim0 lo16
    g1[2] = (int)((((unsigned)H_DIM >> 16) & 0xFFFFu) |
                  (((unsigned)V_DIM & 0xFFFFu) << 16));  // dim0 hi | dim1 lo
    g1[3] = (int)((((unsigned)V_DIM >> 16) & 0xFFFFu) |
                  ((unsigned)KTILE << 16));              // dim1 hi | tile_dim0=512
    g1[4] = 16;                                          // tile_dim1 = 16 rows
    g1[5] = H_DIM;                                       // tensor_dim0_stride = 1024
    int32x4 z = {};
#if TDM_6ARG
    int32x8 z8 = {};
    __builtin_amdgcn_tensor_load_to_lds(g0, g1, z, z, z8, 0);
#else
    __builtin_amdgcn_tensor_load_to_lds(g0, g1, z, z, 0);
#endif
}

// ---------------------------------------------------------------------------
// Utility kernels
// ---------------------------------------------------------------------------
__global__ void gather_emb_bf16_kernel(const int* __restrict__ ids,
                                       const float* __restrict__ emb,
                                       bf16* __restrict__ out, int rows) {
    int idx = blockIdx.x * blockDim.x + threadIdx.x;
    if (idx >= rows * E_DIM) return;
    int r = idx / E_DIM;
    int c = idx - r * E_DIM;
    out[idx] = (bf16)emb[(size_t)ids[r] * E_DIM + c];
}

__global__ void f32_to_bf16_kernel(const float* __restrict__ in,
                                   bf16* __restrict__ out, size_t n) {
    size_t idx = (size_t)blockIdx.x * blockDim.x + threadIdx.x;
    if (idx < n) out[idx] = (bf16)in[idx];
}

__global__ void add_vec_kernel(const float* __restrict__ a,
                               const float* __restrict__ b,
                               float* __restrict__ out, int n) {
    int i = blockIdx.x * blockDim.x + threadIdx.x;
    if (i < n) out[i] = a[i] + b[i];
}

__global__ void zero_u32_kernel(unsigned* __restrict__ p, int n) {
    int i = blockIdx.x * blockDim.x + threadIdx.x;
    if (i < n) p[i] = 0u;
}

__global__ void copy_f32_kernel(const float* __restrict__ src,
                                float* __restrict__ dst, int n) {
    int i = blockIdx.x * blockDim.x + threadIdx.x;
    if (i < n) dst[i] = src[i];
}

__global__ void build_projin_kernel(const bf16* __restrict__ dec,
                                    const float* __restrict__ ctx,
                                    bf16* __restrict__ projin) {
    int idx = blockIdx.x * blockDim.x + threadIdx.x;
    if (idx >= T_LEN * H_DIM) return;
    int t = idx >> 10;
    int k = idx & (H_DIM - 1);
    projin[(size_t)t * (2 * H_DIM) + k]         = dec[idx];
    projin[(size_t)t * (2 * H_DIM) + H_DIM + k] = (bf16)ctx[idx];
}

// ---------------------------------------------------------------------------
// Recurrent core: h_t = tanh(xb[t] + Whh h_{t-1}), xb precomputed by WMMA GEMM.
// 16 co-resident blocks (== max workgroup-cluster size). Step sync:
//  - opportunistic cluster barrier (S_NOP when not dispatched as a cluster)
//  - monotone arrive counter (atomic add release / acquire spin) for guarantee
// Ping-pong h buffers; deterministic fixed-order reductions.
// ---------------------------------------------------------------------------
__global__ __launch_bounds__(1024) void rnn_rec_kernel(
        const float* __restrict__ xb,    // [T,H]
        const float* __restrict__ Whh,   // [H,H]
        float* __restrict__ out,         // [T,H]
        float* __restrict__ h0,          // ping buffer (holds initial h)
        float* __restrict__ h1,          // pong buffer
        unsigned* __restrict__ ctr, int Tn) {
    __shared__ __align__(16) float hsh[H_DIM];
    __shared__ float red[64][16];
    int tid = threadIdx.x;
    int row = tid >> 4;                       // 0..63
    int seg = tid & 15;                       // 16 segments of 64
    int gro = blockIdx.x * 64 + row;          // global output row
    const float4* w4 = (const float4*)(Whh + (size_t)gro * H_DIM + seg * 64);
    float* hb[2] = { h0, h1 };
    for (int t = 0; t < Tn; ++t) {
        hsh[tid] = hb[t & 1][tid];            // stage h (blockDim == H_DIM)
        __syncthreads();
        const float4* hv = (const float4*)(hsh + seg * 64);
        float acc = 0.0f;
        #pragma unroll
        for (int k = 0; k < 16; ++k) {
            float4 w = w4[k], v = hv[k];
            acc += w.x * v.x + w.y * v.y + w.z * v.z + w.w * v.w;
        }
        red[row][seg] = acc;
        __syncthreads();
        if (seg == 0) {
            float s = 0.0f;
            #pragma unroll
            for (int k = 0; k < 16; ++k) s += red[row][k];
            float hn = tanhf(s + xb[(size_t)t * H_DIM + gro]);
            out[(size_t)t * H_DIM + gro] = hn;
            hb[(t + 1) & 1][gro] = hn;
            if (t + 1 < Tn) __builtin_prefetch(&xb[(size_t)(t + 1) * H_DIM + gro], 0, 0);
        }
        __threadfence();
        __syncthreads();
        if (tid == 0) {
            // Fast path if the 16 WGs were dispatched as one cluster; NOP otherwise.
            __builtin_amdgcn_s_cluster_barrier();
            __hip_atomic_fetch_add(ctr, 1u, __ATOMIC_RELEASE, __HIP_MEMORY_SCOPE_AGENT);
            unsigned target = (unsigned)RNN_BLOCKS * (unsigned)(t + 1);
            while (__hip_atomic_load(ctr, __ATOMIC_ACQUIRE, __HIP_MEMORY_SCOPE_AGENT) < target)
                __builtin_amdgcn_s_sleep(1);
        }
        __syncthreads();
    }
}

// ---------------------------------------------------------------------------
// Generic bf16 WMMA GEMM: C[M,N](f32) = A[M,K] * op(B) (+ bias[n])
//   BT==1 : B[k,n] = Bsrc[n*ldb + k]    BT==0 : B[k,n] = Bsrc[k*ldb + n]
// ---------------------------------------------------------------------------
template <int BT>
__global__ __launch_bounds__(256) void gemm_bf16_wmma(
        const bf16* __restrict__ A, const bf16* __restrict__ B,
        float* __restrict__ C, const float* __restrict__ bias,
        int M, int N, int K, int lda, int ldb, int ldc) {
    int lane = threadIdx.x & 31;
    int wave = threadIdx.x >> 5;
    int tile = blockIdx.x * (blockDim.x >> 5) + wave;
    int tilesN = N >> 4;
    if (tile >= (M >> 4) * tilesN) return;        // wave-uniform
    int m0 = (tile / tilesN) << 4;
    int n0 = (tile % tilesN) << 4;
    int lm = lane & 15;
    int half = lane >> 4;
    v8f c = {};
    for (int k0 = 0; k0 < K; k0 += 32) {
        v16bf a, b;
        #pragma unroll
        for (int e = 0; e < 16; ++e) {
            int k = k0 + frag_koff(e, half);
            a[e] = A[(size_t)(m0 + lm) * lda + k];
            if (BT) b[e] = B[(size_t)(n0 + lm) * ldb + k];
            else    b[e] = B[(size_t)k * ldb + (n0 + lm)];
        }
        c = __builtin_amdgcn_wmma_f32_16x16x32_bf16(false, a, false, b,
                                                    (short)0, c, false, false);
    }
    float bs = bias ? bias[n0 + lm] : 0.0f;
    #pragma unroll
    for (int r = 0; r < 8; ++r)
        C[(size_t)(m0 + r + (half << 3)) * ldc + (n0 + lm)] = c[r] + bs;
}

// ---------------------------------------------------------------------------
// Row softmax: aff (f32) -> attn (bf16)
// ---------------------------------------------------------------------------
__global__ void softmax_rows_kernel(const float* __restrict__ aff,
                                    bf16* __restrict__ attn, int ncol) {
    __shared__ float red[256];
    int row = blockIdx.x;
    const float* a = aff + (size_t)row * ncol;
    float mx = -3.4e38f;
    for (int j = threadIdx.x; j < ncol; j += blockDim.x) mx = fmaxf(mx, a[j]);
    red[threadIdx.x] = mx; __syncthreads();
    for (int s = blockDim.x >> 1; s > 0; s >>= 1) {
        if (threadIdx.x < s) red[threadIdx.x] = fmaxf(red[threadIdx.x], red[threadIdx.x + s]);
        __syncthreads();
    }
    mx = red[0]; __syncthreads();
    float sm = 0.0f;
    for (int j = threadIdx.x; j < ncol; j += blockDim.x) sm += __expf(a[j] - mx);
    red[threadIdx.x] = sm; __syncthreads();
    for (int s = blockDim.x >> 1; s > 0; s >>= 1) {
        if (threadIdx.x < s) red[threadIdx.x] += red[threadIdx.x + s];
        __syncthreads();
    }
    float inv = 1.0f / red[0];
    bf16* o = attn + (size_t)row * ncol;
    for (int j = threadIdx.x; j < ncol; j += blockDim.x)
        o[j] = (bf16)(__expf(a[j] - mx) * inv);
}

// ---------------------------------------------------------------------------
// Fused vocab GEMM + online log-softmax partials.
// Grid: 16 row-groups x NCHUNK V-chunks; block = 8 waves, wave w owns rows
// (rg*8+w)*16..+16. W tiles (16 x KTILE bf16) triple-buffered in LDS:
//   even tiles (kh=0): TDM tensor_load_to_lds (wave 0, TENSORcnt)
//   odd  tiles (kh=1): global_load_async_to_lds_b128 (all threads, ASYNCcnt)
// Prefetch distance 1; tiles alternate mechanisms, so exactly one tile is
// outstanding per counter at wait time -> wait-to-zero is precise.
// ---------------------------------------------------------------------------
__global__ __launch_bounds__(256) void scores_loss_kernel(
        const bf16* __restrict__ A,     // proj bf16 [T,H]
        const bf16* __restrict__ W,     // out_W bf16 [V,H]
        const float* __restrict__ out_b,
        const int* __restrict__ target,
        float4* __restrict__ partials)  // [T/16][NCHUNK][16]
{
    __shared__ bf16 wbuf[3][16][WROW];
    __shared__ float stile[8][16][16];
    __shared__ float smax[8][16], ssum[8][16], stgt[8][16];

    int tid = threadIdx.x;
    int lane = tid & 31;
    int wave = tid >> 5;
    int lm = lane & 15;
    int half = lane >> 4;
    int rg = blockIdx.x / NCHUNK;
    int ch = blockIdx.x % NCHUNK;
    int rowTile = rg * 8 + wave;
    int m0 = rowTile << 4;

    if (lane < 16) {
        smax[wave][lane] = -3.4e38f;
        ssum[wave][lane] = 0.0f;
        stgt[wave][lane] = 0.0f;
    }

    unsigned ldsbase = (unsigned)(uintptr_t)(&wbuf[0][0][0]);
    const unsigned bufBytes = 16 * WROW * 2;

    // stage tile idx -> buffer idx%3
    auto stage = [&](int idx) {
        int nt2 = idx >> 1, kh2 = idx & 1;
        int n0s = ch * VCHUNK + nt2 * 16;
        unsigned b = ldsbase + (unsigned)(idx % 3) * bufBytes;
        if (kh2 == 0) {
            if (wave == 0) tdm_load_tile(W + (size_t)n0s * H_DIM, b);
        } else {
            #pragma unroll
            for (int j = 0; j < 4; ++j) {
                int cid = tid + 256 * j;          // 0..1023
                int row = cid >> 6;               // 0..15
                int col = cid & 63;               // 16B chunk within row
                const bf16* g = W + (size_t)(n0s + row) * H_DIM + KTILE + col * 8;
                async_copy_b128(b + (unsigned)(row * (WROW * 2) + col * 16), g);
            }
        }
    };

    stage(0);
    for (int nt = 0; nt < NTILES; ++nt) {
        int n0 = ch * VCHUNK + nt * 16;
        v8f c = {};
        for (int kh = 0; kh < 2; ++kh) {
            int idx = nt * 2 + kh;
            if (idx + 1 < 2 * NTILES) stage(idx + 1);
            if (kh == 0) __builtin_amdgcn_s_wait_tensorcnt(0);  // even tile = TDM
            else         wait_async_0();                        // odd tile = async
            __syncthreads();
            const bf16* wt = &wbuf[idx % 3][0][0];
            for (int k0 = 0; k0 < KTILE; k0 += 32) {
                v16bf a, bfr;
                #pragma unroll
                for (int e = 0; e < 16; ++e) {
                    int kl = k0 + frag_koff(e, half);
                    a[e]   = A[(size_t)(m0 + lm) * H_DIM + kh * KTILE + kl];
                    bfr[e] = wt[lm * WROW + kl];
                }
                c = __builtin_amdgcn_wmma_f32_16x16x32_bf16(false, a, false, bfr,
                                                            (short)0, c, false, false);
            }
        }
        // bias + per-row online log-softmax update (wave-private LDS tile)
        float bs = out_b[n0 + lm];
        #pragma unroll
        for (int r = 0; r < 8; ++r) stile[wave][r + (half << 3)][lm] = c[r] + bs;
        if (lane < 16) {
            int t = m0 + lane;
            int cls = (t < T_LEN - 1) ? target[t + 1] : -1;
            float mx = smax[wave][lane], sm = ssum[wave][lane];
            #pragma unroll
            for (int j = 0; j < 16; ++j) {
                float s = stile[wave][lane][j];
                if (s > mx) { sm *= __expf(mx - s); mx = s; }
                sm += __expf(s - mx);
                if (n0 + j == cls) stgt[wave][lane] = s;
            }
            smax[wave][lane] = mx; ssum[wave][lane] = sm;
        }
    }
    if (lane < 16) {
        partials[((size_t)rowTile * NCHUNK + ch) * 16 + lane] =
            make_float4(smax[wave][lane], ssum[wave][lane], stgt[wave][lane], 0.0f);
    }
}

__global__ void combine_loss_kernel(const float4* __restrict__ partials,
                                    const int* __restrict__ target,
                                    float* __restrict__ rowLoss) {
    int t = blockIdx.x * blockDim.x + threadIdx.x;
    if (t >= T_LEN) return;
    int rt = t >> 4, lr = t & 15;
    int cls = (t < T_LEN - 1) ? target[t + 1] : 0;
    int owner = cls / VCHUNK;
    float mx = -3.4e38f, sm = 0.0f, tg = 0.0f;
    for (int ch = 0; ch < NCHUNK; ++ch) {
        float4 p = partials[((size_t)rt * NCHUNK + ch) * 16 + lr];
        float nm = fmaxf(mx, p.x);
        sm = sm * __expf(mx - nm) + p.y * __expf(p.x - nm);
        mx = nm;
        if (ch == owner) tg = p.z;
    }
    rowLoss[t] = (t < T_LEN - 1) ? (logf(sm) + mx - tg) : 0.0f;
}

__global__ void reduce_loss_kernel(const float* __restrict__ rowLoss,
                                   float* __restrict__ out) {
    __shared__ float red[256];
    float s = 0.0f;
    for (int i = threadIdx.x; i < T_LEN; i += 256) s += rowLoss[i];   // fixed order
    red[threadIdx.x] = s; __syncthreads();
    for (int w = 128; w > 0; w >>= 1) {
        if (threadIdx.x < w) red[threadIdx.x] += red[threadIdx.x + w];
        __syncthreads();
    }
    if (threadIdx.x == 0) out[0] = red[0];
}

// ---------------------------------------------------------------------------
// Host side
// ---------------------------------------------------------------------------
extern "C" void kernel_launch(void* const* d_in, const int* in_sizes, int n_in,
                              void* d_out, int out_size, void* d_ws, size_t ws_size,
                              hipStream_t stream) {
    const int*   source  = (const int*)d_in[0];
    const int*   target  = (const int*)d_in[1];
    const float* emb     = (const float*)d_in[2];
    const float* enc_Wih = (const float*)d_in[3];
    const float* enc_bih = (const float*)d_in[4];
    const float* enc_Whh = (const float*)d_in[5];
    const float* enc_bhh = (const float*)d_in[6];
    const float* dec_Wih = (const float*)d_in[7];
    const float* dec_bih = (const float*)d_in[8];
    const float* dec_Whh = (const float*)d_in[9];
    const float* dec_bhh = (const float*)d_in[10];
    const float* proj_W  = (const float*)d_in[11];
    const float* proj_b  = (const float*)d_in[12];
    const float* out_W   = (const float*)d_in[13];
    const float* out_b   = (const float*)d_in[14];
    float* loss_out = (float*)d_out;

    char* ws = (char*)d_ws;
    size_t off = 0;
    auto take = [&](size_t bytes) -> void* {
        void* p = ws + off;
        off = (off + bytes + 255) & ~(size_t)255;
        return p;
    };

    bf16*  srcE_bf  = (bf16*)take((size_t)S_LEN * E_DIM * 2);
    bf16*  tgtE_bf  = (bf16*)take((size_t)T_LEN * E_DIM * 2);
    bf16*  eWih_bf  = (bf16*)take((size_t)H_DIM * E_DIM * 2);
    bf16*  dWih_bf  = (bf16*)take((size_t)H_DIM * E_DIM * 2);
    float* ebias    = (float*)take((size_t)H_DIM * 4);
    float* dbias    = (float*)take((size_t)H_DIM * 4);
    float* xbE      = (float*)take((size_t)S_LEN * H_DIM * 4);
    float* xbD      = (float*)take((size_t)T_LEN * H_DIM * 4);
    float* hE0      = (float*)take((size_t)H_DIM * 4);
    float* hE1      = (float*)take((size_t)H_DIM * 4);
    float* hD0      = (float*)take((size_t)H_DIM * 4);
    float* hD1      = (float*)take((size_t)H_DIM * 4);
    unsigned* ctrs  = (unsigned*)take(2 * 4);
    float* enc_f32  = (float*)take((size_t)S_LEN * H_DIM * 4);
    float* dec_f32  = (float*)take((size_t)T_LEN * H_DIM * 4);
    bf16*  enc_bf   = (bf16*)take((size_t)S_LEN * H_DIM * 2);
    bf16*  dec_bf   = (bf16*)take((size_t)T_LEN * H_DIM * 2);
    float* aff      = (float*)take((size_t)T_LEN * S_LEN * 4);
    bf16*  attn_bf  = (bf16*)take((size_t)T_LEN * S_LEN * 2);
    float* ctx_f32  = (float*)take((size_t)T_LEN * H_DIM * 4);
    bf16*  projin   = (bf16*)take((size_t)T_LEN * 2 * H_DIM * 2);
    float* proj_f32 = (float*)take((size_t)T_LEN * H_DIM * 4);
    bf16*  proj_bf  = (bf16*)take((size_t)T_LEN * H_DIM * 2);
    bf16*  pW_bf    = (bf16*)take((size_t)H_DIM * 2 * H_DIM * 2);
    bf16*  oW_bf    = (bf16*)take((size_t)V_DIM * H_DIM * 2);
    float4* parts   = (float4*)take((size_t)(T_LEN / 16) * NCHUNK * 16 * 16);
    float* rowLoss  = (float*)take((size_t)T_LEN * 4);

    auto cvt = [&](const float* in, bf16* out, size_t n) {
        f32_to_bf16_kernel<<<(unsigned)((n + 255) / 256), 256, 0, stream>>>(in, out, n);
    };

    // embeddings (direct to bf16) + weight conversions
    gather_emb_bf16_kernel<<<(S_LEN * E_DIM + 255) / 256, 256, 0, stream>>>(source, emb, srcE_bf, S_LEN);
    gather_emb_bf16_kernel<<<(T_LEN * E_DIM + 255) / 256, 256, 0, stream>>>(target, emb, tgtE_bf, T_LEN);
    cvt(enc_Wih, eWih_bf, (size_t)H_DIM * E_DIM);
    cvt(dec_Wih, dWih_bf, (size_t)H_DIM * E_DIM);
    cvt(proj_W, pW_bf, (size_t)H_DIM * 2 * H_DIM);
    cvt(out_W, oW_bf, (size_t)V_DIM * H_DIM);
    add_vec_kernel<<<(H_DIM + 255) / 256, 256, 0, stream>>>(enc_bih, enc_bhh, ebias, H_DIM);
    add_vec_kernel<<<(H_DIM + 255) / 256, 256, 0, stream>>>(dec_bih, dec_bhh, dbias, H_DIM);

    // input halves of both RNNs as parallel WMMA GEMMs: xb = X * Wih^T + (bih+bhh)
    {
        int tiles = (S_LEN / 16) * (H_DIM / 16);
        gemm_bf16_wmma<1><<<tiles / 8, 256, 0, stream>>>(
            srcE_bf, eWih_bf, xbE, ebias, S_LEN, H_DIM, E_DIM, E_DIM, E_DIM, H_DIM);
        gemm_bf16_wmma<1><<<tiles / 8, 256, 0, stream>>>(
            tgtE_bf, dWih_bf, xbD, dbias, T_LEN, H_DIM, E_DIM, E_DIM, E_DIM, H_DIM);
    }

    // serial recurrences across 16 WGPs with global step barrier
    zero_u32_kernel<<<1, 32, 0, stream>>>(ctrs, 2);
    zero_u32_kernel<<<(H_DIM + 255) / 256, 256, 0, stream>>>((unsigned*)hE0, H_DIM);
    rnn_rec_kernel<<<RNN_BLOCKS, 1024, 0, stream>>>(xbE, enc_Whh, enc_f32, hE0, hE1, ctrs + 0, S_LEN);
    copy_f32_kernel<<<(H_DIM + 255) / 256, 256, 0, stream>>>(hE0, hD0, H_DIM);  // final h (even T) in hE0
    rnn_rec_kernel<<<RNN_BLOCKS, 1024, 0, stream>>>(xbD, dec_Whh, dec_f32, hD0, hD1, ctrs + 1, T_LEN);

    cvt(enc_f32, enc_bf, (size_t)S_LEN * H_DIM);
    cvt(dec_f32, dec_bf, (size_t)T_LEN * H_DIM);

    // attention: aff = dec * enc^T ; softmax ; ctx = attn * enc
    {
        int tiles = (T_LEN / 16) * (S_LEN / 16);
        gemm_bf16_wmma<1><<<tiles / 8, 256, 0, stream>>>(
            dec_bf, enc_bf, aff, nullptr, T_LEN, S_LEN, H_DIM, H_DIM, H_DIM, S_LEN);
    }
    softmax_rows_kernel<<<T_LEN, 256, 0, stream>>>(aff, attn_bf, S_LEN);
    {
        int tiles = (T_LEN / 16) * (H_DIM / 16);
        gemm_bf16_wmma<0><<<tiles / 8, 256, 0, stream>>>(
            attn_bf, enc_bf, ctx_f32, nullptr, T_LEN, H_DIM, S_LEN, S_LEN, H_DIM, H_DIM);
    }

    // projection
    build_projin_kernel<<<(T_LEN * H_DIM + 255) / 256, 256, 0, stream>>>(dec_bf, ctx_f32, projin);
    {
        int tiles = (T_LEN / 16) * (H_DIM / 16);
        gemm_bf16_wmma<1><<<tiles / 8, 256, 0, stream>>>(
            projin, pW_bf, proj_f32, proj_b, T_LEN, H_DIM, 2 * H_DIM, 2 * H_DIM, 2 * H_DIM, H_DIM);
    }
    cvt(proj_f32, proj_bf, (size_t)T_LEN * H_DIM);

    // fused vocab GEMM (TDM + async LDS staged) + online softmax partials + merge
    scores_loss_kernel<<<(T_LEN / 128) * NCHUNK, 256, 0, stream>>>(
        proj_bf, oW_bf, out_b, target, parts);
    combine_loss_kernel<<<(T_LEN + 255) / 256, 256, 0, stream>>>(parts, target, rowLoss);
    reduce_loss_kernel<<<1, 256, 0, stream>>>(rowLoss, loss_out);
}
